// TG_MSA_17703855194186
// MI455X (gfx1250) — matrix operand order repaired and checked by reading, not compiled, not run on hardware
//
#include <hip/hip_runtime.h>
#include <stdint.h>

// ---------------------------------------------------------------------------
// Problem constants (reference: B=1, N=8192, DIM=512, HEADS=8, DIM_HEAD=64)
// ---------------------------------------------------------------------------
#define NSEQ   8192
#define DIMC   512
#define NHEADS 8
#define DHEAD  64

typedef __attribute__((ext_vector_type(16))) __bf16 bf16x16;
typedef __attribute__((ext_vector_type(8)))  float  f32x8;

struct alignas(16) U128 { unsigned x, y, z, w; };   // POD 16B chunk (union-safe)

// fp32 -> bf16, round-to-nearest-even (inputs are finite; skip NaN path)
__device__ __forceinline__ unsigned short f2bf(float x) {
  union { float f; unsigned u; } v; v.f = x;
  unsigned r = v.u + 0x7FFFu + ((v.u >> 16) & 1u);
  return (unsigned short)(r >> 16);
}

// ---------------------------------------------------------------------------
// Elementwise / conversion kernels
// ---------------------------------------------------------------------------
__global__ void cvt_f32_bf16(const float* __restrict__ in,
                             unsigned short* __restrict__ out, long long n) {
  long long i = ((long long)blockIdx.x * blockDim.x + threadIdx.x) * 4;
  if (i + 4 <= n) {
    float4 v = *(const float4*)(in + i);
    out[i + 0] = f2bf(v.x); out[i + 1] = f2bf(v.y);
    out[i + 2] = f2bf(v.z); out[i + 3] = f2bf(v.w);
  }
}

// out[c, r] = bf16(in[r, c])   (in: fp32 [R,C] row-major -> out: bf16 [C,R])
__global__ void transpose_cvt(const float* __restrict__ in,
                              unsigned short* __restrict__ out, int R, int C) {
  __shared__ float tile[32][33];
  int bx = blockIdx.x * 32;            // column base (input)
  int by = blockIdx.y * 32;            // row base (input)
  int tx = threadIdx.x, ty = threadIdx.y;   // 32 x 8
  #pragma unroll
  for (int i = ty; i < 32; i += 8) {
    int r = by + i, c = bx + tx;
    tile[i][tx] = (r < R && c < C) ? in[(size_t)r * C + c] : 0.f;
  }
  __syncthreads();
  #pragma unroll
  for (int i = ty; i < 32; i += 8) {
    int c = bx + i, r = by + tx;
    if (c < C && r < R) out[(size_t)c * R + r] = f2bf(tile[tx][i]);
  }
}

// guided = bf16(gfull * v), vb = bf16(v)
__global__ void guided_mul(const float* __restrict__ g, const float* __restrict__ v,
                           unsigned short* __restrict__ gb,
                           unsigned short* __restrict__ vb, int n) {
  int i = blockIdx.x * 256 + threadIdx.x;
  if (i < n) { float vv = v[i]; gb[i] = f2bf(g[i] * vv); vb[i] = f2bf(vv); }
}

// Column L2 norms of q and k (columns of [R,C] fp32), clamped to eps.
__global__ void col_norms(const float* __restrict__ q, const float* __restrict__ k,
                          float* __restrict__ qn, float* __restrict__ kn,
                          int R, int C) {
  __shared__ float red[256];
  int idx = blockIdx.x;
  int col = (idx < C) ? idx : idx - C;
  const float* src = (idx < C) ? q : k;
  float s = 0.f;
  for (int r = threadIdx.x; r < R; r += 256) {
    float x = src[(size_t)r * C + col];
    s += x * x;
  }
  red[threadIdx.x] = s;
  __syncthreads();
  for (int off = 128; off > 0; off >>= 1) {
    if ((int)threadIdx.x < off) red[threadIdx.x] += red[threadIdx.x + off];
    __syncthreads();
  }
  if (threadIdx.x == 0) {
    float nv = fmaxf(sqrtf(red[0]), 1e-12f);
    ((idx < C) ? qn : kn)[col] = nv;
  }
}

// attn[h,d,e] = softmax_e( S[h,d,e]*rescale[h] / (kn[h64+d]*qn[h64+e]) ) -> bf16
__global__ void softmax_attn(const float* __restrict__ S,
                             const float* __restrict__ qn, const float* __restrict__ kn,
                             const float* __restrict__ rescale,
                             unsigned short* __restrict__ Ab) {
  __shared__ float buf[64];
  int h = blockIdx.x >> 6, d = blockIdx.x & 63, e = threadIdx.x;   // 64 threads
  float kd = kn[(h << 6) + d];
  float x = S[(size_t)((h << 6) + d) * 64 + e] * rescale[h] / (kd * qn[(h << 6) + e]);
  buf[e] = x;
  __syncthreads();
  for (int off = 32; off > 0; off >>= 1) {
    if (e < off) buf[e] = fmaxf(buf[e], buf[e + off]);
    __syncthreads();
  }
  float mx = buf[0];
  __syncthreads();
  float ex = __expf(x - mx);
  buf[e] = ex;
  __syncthreads();
  for (int off = 32; off > 0; off >>= 1) {
    if (e < off) buf[e] += buf[e + off];
    __syncthreads();
  }
  float sum = buf[0];
  Ab[(size_t)((h << 6) + d) * 64 + e] = f2bf(ex / sum);
}

// h1b = bf16( gelu_exact(h + b1) )
__global__ void bias_gelu_cvt(const float* __restrict__ h, const float* __restrict__ b,
                              unsigned short* __restrict__ out, int n) {
  int i = blockIdx.x * 256 + threadIdx.x;
  if (i < n) {
    float x = h[i] + b[i & (DIMC - 1)];
    out[i] = f2bf(0.5f * x * (1.f + erff(x * 0.70710678118654752440f)));
  }
}

// out = outc + outp + bp[c] + b2[c]
__global__ void final_add(const float* __restrict__ oc, const float* __restrict__ op,
                          const float* __restrict__ bp, const float* __restrict__ b2,
                          float* __restrict__ out, int n) {
  int i = blockIdx.x * 256 + threadIdx.x;
  if (i < n) {
    int c = i & (DIMC - 1);
    out[i] = oc[i] + op[i] + bp[c] + b2[c];
  }
}

// ---------------------------------------------------------------------------
// WMMA bf16 GEMM:  C[M,N] (f32) = X[M,K] (bf16,row-major) @ W[N,K]^T (bf16,row-major)
// Block = 256 threads = 8 waves; block tile 128x128; each wave 32x64 (2x4 frags).
// Fragments loaded straight from global, A and B in the same contiguous k order:
//   A (16x32, lane=row M):  lane<16 holds K{0..15},       lane>=16 K{16..31}
//   B (32x16, lane=col N):  lane<16 holds K{0..15},       lane>=16 K{16..31}
// Interior tiles take an unguarded, double-buffered fast path (K % 64 == 0 in
// all calls); edge tiles (only the tiny 64-wide GEMMs) take a guarded path.
// ---------------------------------------------------------------------------
__device__ __forceinline__ bf16x16 mkfrag(U128 lo, U128 hi) {
  union { U128 q[2]; bf16x16 v; } f;
  f.q[0] = lo; f.q[1] = hi;
  return f.v;
}

struct FragBuf {
  U128 a0l, a0h, a1l, a1h;
  U128 b0l, b0h, b1l, b1h, b2l, b2h, b3l, b3h;
};

__device__ __forceinline__ void load12(FragBuf& f,
    const unsigned short* __restrict__ pa0, const unsigned short* __restrict__ pa1,
    const unsigned short* __restrict__ pb0, const unsigned short* __restrict__ pb1,
    const unsigned short* __restrict__ pb2, const unsigned short* __restrict__ pb3,
    int k) {
  f.a0l = *(const U128*)(pa0 + k); f.a0h = *(const U128*)(pa0 + k + 8);
  f.a1l = *(const U128*)(pa1 + k); f.a1h = *(const U128*)(pa1 + k + 8);
  f.b0l = *(const U128*)(pb0 + k); f.b0h = *(const U128*)(pb0 + k + 8);
  f.b1l = *(const U128*)(pb1 + k); f.b1h = *(const U128*)(pb1 + k + 8);
  f.b2l = *(const U128*)(pb2 + k); f.b2h = *(const U128*)(pb2 + k + 8);
  f.b3l = *(const U128*)(pb3 + k); f.b3h = *(const U128*)(pb3 + k + 8);
}

__device__ __forceinline__ void wmma8(const FragBuf& f, f32x8 (&acc)[2][4]) {
  bf16x16 a0 = mkfrag(f.a0l, f.a0h);
  bf16x16 a1 = mkfrag(f.a1l, f.a1h);
  bf16x16 b0 = mkfrag(f.b0l, f.b0h);
  bf16x16 b1 = mkfrag(f.b1l, f.b1h);
  bf16x16 b2 = mkfrag(f.b2l, f.b2h);
  bf16x16 b3 = mkfrag(f.b3l, f.b3h);
  acc[0][0] = __builtin_amdgcn_wmma_f32_16x16x32_bf16(false, a0, false, b0, (short)0, acc[0][0], false, false);
  acc[0][1] = __builtin_amdgcn_wmma_f32_16x16x32_bf16(false, a0, false, b1, (short)0, acc[0][1], false, false);
  acc[0][2] = __builtin_amdgcn_wmma_f32_16x16x32_bf16(false, a0, false, b2, (short)0, acc[0][2], false, false);
  acc[0][3] = __builtin_amdgcn_wmma_f32_16x16x32_bf16(false, a0, false, b3, (short)0, acc[0][3], false, false);
  acc[1][0] = __builtin_amdgcn_wmma_f32_16x16x32_bf16(false, a1, false, b0, (short)0, acc[1][0], false, false);
  acc[1][1] = __builtin_amdgcn_wmma_f32_16x16x32_bf16(false, a1, false, b1, (short)0, acc[1][1], false, false);
  acc[1][2] = __builtin_amdgcn_wmma_f32_16x16x32_bf16(false, a1, false, b2, (short)0, acc[1][2], false, false);
  acc[1][3] = __builtin_amdgcn_wmma_f32_16x16x32_bf16(false, a1, false, b3, (short)0, acc[1][3], false, false);
}

// Guarded fragment loads for edge tiles
__device__ __forceinline__ bf16x16 load_a_frag(const unsigned short* __restrict__ p,
                                               int ld, int rbase, int kbase,
                                               int rmax, int lane) {
  union { U128 q[2]; unsigned u[8]; bf16x16 v; } f;
  int r = rbase + (lane & 15);
  if (r < rmax) {
    const unsigned short* s = p + (size_t)r * ld + kbase + ((lane >> 4) << 4);
    f.q[0] = *(const U128*)s;
    f.q[1] = *(const U128*)(s + 8);
  } else {
    #pragma unroll
    for (int i = 0; i < 8; ++i) f.u[i] = 0u;
  }
  return f.v;
}

__device__ __forceinline__ bf16x16 load_b_frag(const unsigned short* __restrict__ p,
                                               int ld, int nbase, int kbase,
                                               int nmax, int lane) {
  union { U128 q[2]; unsigned u[8]; bf16x16 v; } f;
  int r = nbase + (lane & 15);
  if (r < nmax) {
    const unsigned short* s = p + (size_t)r * ld + kbase + ((lane >> 4) << 4);
    f.q[0] = *(const U128*)s;
    f.q[1] = *(const U128*)(s + 8);
  } else {
    #pragma unroll
    for (int i = 0; i < 8; ++i) f.u[i] = 0u;
  }
  return f.v;
}

__global__ __launch_bounds__(256) void gemm_bf16_nt(
    const unsigned short* __restrict__ X, const unsigned short* __restrict__ Wm,
    float* __restrict__ C, int M, int N, int K, int ldx, int ldw, int ldc,
    long long bsX, long long bsW, long long bsC) {
  X  += (long long)blockIdx.z * bsX;
  Wm += (long long)blockIdx.z * bsW;
  C  += (long long)blockIdx.z * bsC;

  const int lane  = threadIdx.x & 31;
  const int wave  = threadIdx.x >> 5;
  const int rbase = blockIdx.y * 128 + (wave & 3) * 32;   // M
  const int cbase = blockIdx.x * 128 + (wave >> 2) * 64;  // N

  f32x8 acc[2][4];
  #pragma unroll
  for (int i = 0; i < 2; ++i)
    #pragma unroll
    for (int j = 0; j < 4; ++j)
      #pragma unroll
      for (int t = 0; t < 8; ++t) acc[i][j][t] = 0.f;

  // Wave-uniform interior test: every fragment of this wave's 32x64 tile is in
  // bounds, so the hot loop carries no EXEC juggling and no zero-refills.
  const bool interior = (rbase + 32 <= M) && (cbase + 64 <= N);

  if (interior) {
    const unsigned short* pa0 = X  + (size_t)(rbase + (lane & 15)) * ldx + ((lane >> 4) << 4);
    const unsigned short* pa1 = pa0 + (size_t)16 * ldx;
    const unsigned short* pb0 = Wm + (size_t)(cbase + (lane & 15)) * ldw + ((lane >> 4) << 4);
    const unsigned short* pb1 = pb0 + (size_t)16 * ldw;
    const unsigned short* pb2 = pb0 + (size_t)32 * ldw;
    const unsigned short* pb3 = pb0 + (size_t)48 * ldw;

    FragBuf f0, f1;
    load12(f0, pa0, pa1, pb0, pb1, pb2, pb3, 0);
    // K % 64 == 0 for every call site (512, 8192, 64).
    for (int k = 0; k < K; k += 64) {
      __builtin_prefetch(pa0 + k + 128, 0, 3);   // speculative, 2 tiles ahead
      __builtin_prefetch(pb0 + k + 128, 0, 3);
      load12(f1, pa0, pa1, pb0, pb1, pb2, pb3, k + 32);
      wmma8(f0, acc);                            // overlaps with f1 loads
      if (k + 64 < K)
        load12(f0, pa0, pa1, pb0, pb1, pb2, pb3, k + 64);
      wmma8(f1, acc);                            // overlaps with f0 loads
    }

    const int rofs = (lane >> 4) << 3;
    const int cofs = lane & 15;
    #pragma unroll
    for (int i = 0; i < 2; ++i)
      #pragma unroll
      for (int j = 0; j < 4; ++j) {
        float* cp = C + (size_t)(rbase + 16 * i + rofs) * ldc + (cbase + 16 * j + cofs);
        #pragma unroll
        for (int t = 0; t < 8; ++t) cp[(size_t)t * ldc] = acc[i][j][t];
      }
  } else {
    // Edge path (used only by the 64-wide GEMMs): per-lane guarded loads.
    for (int k = 0; k < K; k += 32) {
      FragBuf f;
      bf16x16 a0 = load_a_frag(X, ldx, rbase,      k, M, lane);
      bf16x16 a1 = load_a_frag(X, ldx, rbase + 16, k, M, lane);
      bf16x16 b0 = load_b_frag(Wm, ldw, cbase +  0, k, N, lane);
      bf16x16 b1 = load_b_frag(Wm, ldw, cbase + 16, k, N, lane);
      bf16x16 b2 = load_b_frag(Wm, ldw, cbase + 32, k, N, lane);
      bf16x16 b3 = load_b_frag(Wm, ldw, cbase + 48, k, N, lane);
      (void)f;
      acc[0][0] = __builtin_amdgcn_wmma_f32_16x16x32_bf16(false, a0, false, b0, (short)0, acc[0][0], false, false);
      acc[0][1] = __builtin_amdgcn_wmma_f32_16x16x32_bf16(false, a0, false, b1, (short)0, acc[0][1], false, false);
      acc[0][2] = __builtin_amdgcn_wmma_f32_16x16x32_bf16(false, a0, false, b2, (short)0, acc[0][2], false, false);
      acc[0][3] = __builtin_amdgcn_wmma_f32_16x16x32_bf16(false, a0, false, b3, (short)0, acc[0][3], false, false);
      acc[1][0] = __builtin_amdgcn_wmma_f32_16x16x32_bf16(false, a1, false, b0, (short)0, acc[1][0], false, false);
      acc[1][1] = __builtin_amdgcn_wmma_f32_16x16x32_bf16(false, a1, false, b1, (short)0, acc[1][1], false, false);
      acc[1][2] = __builtin_amdgcn_wmma_f32_16x16x32_bf16(false, a1, false, b2, (short)0, acc[1][2], false, false);
      acc[1][3] = __builtin_amdgcn_wmma_f32_16x16x32_bf16(false, a1, false, b3, (short)0, acc[1][3], false, false);
    }
    const int rofs = (lane >> 4) << 3;
    const int cofs = lane & 15;
    #pragma unroll
    for (int i = 0; i < 2; ++i)
      #pragma unroll
      for (int j = 0; j < 4; ++j)
        #pragma unroll
        for (int t = 0; t < 8; ++t) {
          int r = rbase + 16 * i + rofs + t;
          int c = cbase + 16 * j + cofs;
          if (r < M && c < N) C[(size_t)r * ldc + c] = acc[i][j][t];
        }
  }
}

// ---------------------------------------------------------------------------
// Host-side orchestration
// ---------------------------------------------------------------------------
extern "C" void kernel_launch(void* const* d_in, const int* in_sizes, int n_in,
                              void* d_out, int out_size, void* d_ws, size_t ws_size,
                              hipStream_t stream) {
  (void)in_sizes; (void)n_in; (void)out_size; (void)ws_size;
  const float* x_in    = (const float*)d_in[0];
  const float* task_x  = (const float*)d_in[1];
  const float* Ges     = (const float*)d_in[2];
  const float* Wq      = (const float*)d_in[3];
  const float* Wk      = (const float*)d_in[4];
  const float* Wv      = (const float*)d_in[5];
  const float* rescale = (const float*)d_in[6];
  const float* Wp      = (const float*)d_in[7];
  const float* bp      = (const float*)d_in[8];
  const float* W1      = (const float*)d_in[9];
  const float* b1      = (const float*)d_in[10];
  const float* W2      = (const float*)d_in[11];
  const float* b2      = (const float*)d_in[12];
  float* out = (float*)d_out;

  char* ws = (char*)d_ws;
  size_t off = 0;
  auto take = [&](size_t bytes) -> void* {
    void* p = ws + off; off += (bytes + 255) & ~(size_t)255; return p;
  };

  const long long NC  = (long long)NSEQ * DIMC;   // 4.19M
  const long long NN  = (long long)NSEQ * NSEQ;   // 67.1M
  const long long CC  = (long long)DIMC * DIMC;   // 0.26M

  unsigned short* Gb  = (unsigned short*)take(NN * 2);    // bf16 Ges
  unsigned short* xb  = (unsigned short*)take(NC * 2);    // bf16 x_in
  unsigned short* tbT = (unsigned short*)take(NC * 2);    // bf16 task_x^T  [512,8192]
  unsigned short* Wqb = (unsigned short*)take(CC * 2);
  unsigned short* Wkb = (unsigned short*)take(CC * 2);
  unsigned short* Wvb = (unsigned short*)take(CC * 2);
  unsigned short* Wpb = (unsigned short*)take(CC * 2);
  unsigned short* W1b = (unsigned short*)take(CC * 2);
  unsigned short* W2b = (unsigned short*)take(CC * 2);
  float* qf    = (float*)take(NC * 4);
  float* kf    = (float*)take(NC * 4);
  float* vf    = (float*)take(NC * 4);
  float* gfull = (float*)take(NC * 4);
  unsigned short* gub = (unsigned short*)take(NC * 2);    // bf16 guided
  unsigned short* vb  = (unsigned short*)take(NC * 2);    // bf16 v
  unsigned short* qTb = (unsigned short*)take(NC * 2);    // bf16 q^T [512,8192]
  unsigned short* kTb = (unsigned short*)take(NC * 2);    // bf16 k^T [512,8192]
  float* S   = (float*)take((size_t)NHEADS * DHEAD * DHEAD * 4);
  float* qn  = (float*)take(DIMC * 4);
  float* kn  = (float*)take(DIMC * 4);
  unsigned short* Ab = (unsigned short*)take((size_t)NHEADS * DHEAD * DHEAD * 2);
  // Buffer reuse (after last use of original contents):
  float* Y            = gfull;   // Y written after gfull consumed
  unsigned short* Yb  = xb;      // xb dead after q/k/v GEMMs
  float* outc         = qf;      // q dead after qTb + norms
  float* h1           = kf;      // k dead after kTb + norms
  unsigned short* h1b = tbT;     // tbT dead after Ges GEMM
  float* outp         = vf;      // v dead after guided_mul

  auto gemm = [&](const unsigned short* Xp, const unsigned short* Wmp, float* Cp,
                  int M, int N, int K, int ldx, int ldw, int ldc,
                  long long bsX, long long bsW, long long bsC, int batch) {
    dim3 g((unsigned)((N + 127) / 128), (unsigned)((M + 127) / 128), (unsigned)batch);
    gemm_bf16_nt<<<g, 256, 0, stream>>>(Xp, Wmp, Cp, M, N, K, ldx, ldw, ldc,
                                        bsX, bsW, bsC);
  };
  auto cvt = [&](const float* in, unsigned short* o, long long n) {
    cvt_f32_bf16<<<(unsigned)((n / 4 + 255) / 256), 256, 0, stream>>>(in, o, n);
  };

  // 1) fp32 -> bf16 conversions (Ges is the big one: 268MB read, 134MB write)
  cvt(Ges, Gb, NN);
  cvt(x_in, xb, NC);
  cvt(Wq, Wqb, CC); cvt(Wk, Wkb, CC); cvt(Wv, Wvb, CC);
  cvt(Wp, Wpb, CC); cvt(W1, W1b, CC); cvt(W2, W2b, CC);

  // 2) task_x^T (bf16) so Ges@task_x becomes the canonical X@W^T form
  transpose_cvt<<<dim3(DIMC / 32, NSEQ / 32), dim3(32, 8), 0, stream>>>(task_x, tbT, NSEQ, DIMC);

  // 3) q,k,v projections: [8192,512] = xb @ W^T
  gemm(xb, Wqb, qf, NSEQ, DIMC, DIMC, DIMC, DIMC, DIMC, 0, 0, 0, 1);
  gemm(xb, Wkb, kf, NSEQ, DIMC, DIMC, DIMC, DIMC, DIMC, 0, 0, 0, 1);
  gemm(xb, Wvb, vf, NSEQ, DIMC, DIMC, DIMC, DIMC, DIMC, 0, 0, 0, 1);

  // 4) gfull = Ges @ task_x  (dominant GEMM, 8192x512x8192; Gb is L2-resident)
  gemm(Gb, tbT, gfull, NSEQ, DIMC, NSEQ, NSEQ, NSEQ, DIMC, 0, 0, 0, 1);

  // 5) guided = bf16(gfull * v); vb = bf16(v)
  guided_mul<<<(unsigned)((NC + 255) / 256), 256, 0, stream>>>(gfull, vf, gub, vb, (int)NC);

  // 6) q^T, k^T in bf16 for the S GEMM
  transpose_cvt<<<dim3(DIMC / 32, NSEQ / 32), dim3(32, 8), 0, stream>>>(qf, qTb, NSEQ, DIMC);
  transpose_cvt<<<dim3(DIMC / 32, NSEQ / 32), dim3(32, 8), 0, stream>>>(kf, kTb, NSEQ, DIMC);

  // 7) column norms of q and k (fp32, clamped to eps)
  col_norms<<<2 * DIMC, 256, 0, stream>>>(qf, kf, qn, kn, NSEQ, DIMC);

  // 8) S[h] = kT_h @ qT_h^T : batched 64x64 with K=8192
  gemm(kTb, qTb, S, DHEAD, DHEAD, NSEQ, NSEQ, NSEQ, DHEAD,
       (long long)DHEAD * NSEQ, (long long)DHEAD * NSEQ, (long long)DHEAD * DHEAD, NHEADS);

  // 9) attn = softmax(normalized S) -> bf16
  softmax_attn<<<NHEADS * DHEAD, DHEAD, 0, stream>>>(S, qn, kn, rescale, Ab);

  // 10) Y[n, h*64+d] = sum_e guided[n, h*64+e] * attn[h,d,e] : batched 8192x64x64
  gemm(gub, Ab, Y, NSEQ, DHEAD, DHEAD, DIMC, DHEAD, DIMC,
       (long long)DHEAD, (long long)DHEAD * DHEAD, (long long)DHEAD, NHEADS);

  // 11) out_c = Y @ Wp^T
  cvt(Y, Yb, NC);
  gemm(Yb, Wpb, outc, NSEQ, DIMC, DIMC, DIMC, DIMC, DIMC, 0, 0, 0, 1);

  // 12) out_p = gelu(v @ W1^T + b1) @ W2^T
  gemm(vb, W1b, h1, NSEQ, DIMC, DIMC, DIMC, DIMC, DIMC, 0, 0, 0, 1);
  bias_gelu_cvt<<<(unsigned)((NC + 255) / 256), 256, 0, stream>>>(h1, b1, h1b, (int)NC);
  gemm(h1b, W2b, outp, NSEQ, DIMC, DIMC, DIMC, DIMC, DIMC, 0, 0, 0, 1);

  // 13) out = out_c + out_p + bp + b2
  final_add<<<(unsigned)((NC + 255) / 256), 256, 0, stream>>>(outc, outp, bp, b2, out, (int)NC);
}
